// BinaryNormalizedConv2d_33981781246444
// MI455X (gfx1250) — matrix operand order repaired and checked by reading, not compiled, not run on hardware
//
#include <hip/hip_runtime.h>
#include <hip/hip_bf16.h>

typedef __attribute__((ext_vector_type(16))) __bf16 v16bf;
typedef __attribute__((ext_vector_type(8)))  float  v8f;

// Problem shape (fixed by reference)
constexpr int B_   = 32;
constexpr int CIN  = 128;
constexpr int H_   = 56;
constexpr int W_   = 56;
constexpr int COUT = 256;
constexpr int KTOT = CIN * 9;        // K order: k = (kh*3+kw)*128 + ci
constexpr int KCH  = 32;
constexpr int NKC  = KTOT / KCH;     // 36
constexpr int NB   = COUT * H_ * W_; // 802816 per sample
constexpr int PB   = 98;             // 98 * 8192 = 802816

// Padded NHWC x copy: [B][58][58][128] bf16, zero halo (1 px each side)
constexpr int HP = H_ + 2, WP = W_ + 2;
constexpr size_t XCN = (size_t)B_ * HP * WP * CIN;          // elements
constexpr size_t XC_BYTES = XCN * 2 + (64 << 10);           // + tail slack

// ---------------- Kernel 1: scalar means of weight and bias ----------------
__global__ void k_means(const float* __restrict__ w, const float* __restrict__ b,
                        float* __restrict__ misc) {
  __shared__ float sm[1024];
  int t = threadIdx.x;
  float s = 0.f;
  for (int i = t; i < COUT * CIN * 9; i += 1024) s += w[i];
  sm[t] = s; __syncthreads();
  for (int o = 512; o > 0; o >>= 1) { if (t < o) sm[t] += sm[t + o]; __syncthreads(); }
  if (t == 0) misc[0] = sm[0] / (float)(COUT * CIN * 9);
  __syncthreads();
  sm[t] = (t < COUT) ? b[t] : 0.f; __syncthreads();
  for (int o = 512; o > 0; o >>= 1) { if (t < o) sm[t] += sm[t + o]; __syncthreads(); }
  if (t == 0) misc[1] = sm[0] / (float)COUT;
}

// ------- Kernel 2: binarize weight, pack into CDNA5 A-fragment layout -------
// 16-bit A 16x32: lane<16 -> K = hh; lane>=16: K = hh+16 (contiguous, matches B).
__global__ void k_packA(const float* __restrict__ w, const float* __restrict__ misc,
                        __bf16* __restrict__ Apack) {
  int kc = blockIdx.x % NKC;
  int mt = blockIdx.x / NKC;
  int L  = threadIdx.x;            // 32 lanes
  float mw = misc[0];
  int cout = mt * 16 + (L & 15);
  int lo   = (L < 16) ? 0 : 16;
  __bf16 frag[16];
  #pragma unroll
  for (int hh = 0; hh < 16; ++hh) {
    int koff = hh + lo;
    int r  = kc >> 2;                       // kh*3+kw
    int ci = ((kc & 3) << 5) + koff;
    int kh = r / 3, kw2 = r - kh * 3;
    float wv = w[((cout * CIN + ci) * 3 + kh) * 3 + kw2];
    frag[hh] = (__bf16)((wv > mw) ? 1.0f : 0.0f);
  }
  __bf16* dst = Apack + ((size_t)(mt * NKC + kc) * 32 + L) * 16;
  #pragma unroll
  for (int hh = 0; hh < 16; ++hh) dst[hh] = frag[hh];
}

// ---------------- Kernel 3: binarize bias ----------------
__global__ void k_bq(const float* __restrict__ bias, const float* __restrict__ misc,
                     float* __restrict__ bq) {
  int t = threadIdx.x;
  bq[t] = (bias[t] > misc[1]) ? 1.f : 0.f;
}

// ---------------- Kernel 4a: zero the padded xc buffer ----------------
__global__ void k_zero(uint4* __restrict__ p, size_t n16) {
  uint4 z = {0u, 0u, 0u, 0u};
  for (size_t g = (size_t)blockIdx.x * 256 + threadIdx.x; g < n16;
       g += (size_t)gridDim.x * 256) p[g] = z;
}

// ---------------- Kernel 4b: x NCHW f32 -> padded NHWC bf16 ----------------
__global__ void k_xc(const float* __restrict__ x, __bf16* __restrict__ xc) {
  size_t g = (size_t)blockIdx.x * 256 + threadIdx.x;   // over B*H*W*C
  int c = (int)(g & 127);
  size_t r = g >> 7;
  int wq = (int)(r % W_); r /= W_;
  int h  = (int)(r % H_);
  int b  = (int)(r / H_);
  xc[((size_t)(b * HP + h + 1) * WP + (wq + 1)) * CIN + c] =
      (__bf16)x[(((size_t)b * CIN + c) * H_ + h) * W_ + wq];
}

// ---------------- Kernel 5: implicit-GEMM conv via WMMA bf16 ----------------
// Block = (b, row-pair hp, cout-tile mt); 8 waves: h = hp*2+(wv>>2), wseg = (wv&3)*16.
// A slab (36 KB) staged to LDS with async-to-LDS; K-loop fully unrolled with
// immediate-offset loads thanks to the zero-halo layout (branchless).
__global__ __launch_bounds__(256) void k_conv(const __bf16* __restrict__ Apack,
                                              const __bf16* __restrict__ xc,
                                              const float*  __restrict__ bq,
                                              float* __restrict__ out) {
  __shared__ __align__(32) __bf16 Alds[NKC * 512];   // 36 KB

  int mt  = blockIdx.x & 15;
  int tmp = blockIdx.x >> 4;
  int hp  = tmp % 28;
  int b   = tmp / 28;

  // Async cooperative copy: 2304 x b128, 9 per thread (ASYNCcnt-tracked).
  {
    const __bf16* gsrc = Apack + (size_t)mt * NKC * 512;
    unsigned lbase = (unsigned)(uintptr_t)&Alds[0];
    #pragma unroll
    for (int i = 0; i < 9; ++i) {
      int e = threadIdx.x + i * 256;
      const __bf16* gp = gsrc + e * 8;
      unsigned lp = lbase + e * 16;
      asm volatile("global_load_async_to_lds_b128 %0, %1, off"
                   :: "v"(lp), "v"(gp) : "memory");
    }
    asm volatile("s_wait_asynccnt 0" ::: "memory");
  }
  __syncthreads();

  int wv   = threadIdx.x >> 5;
  int lane = threadIdx.x & 31;
  int h    = hp * 2 + (wv >> 2);
  int n    = lane & 15;
  int w    = (wv & 3) * 16 + n;          // may exceed 55 in last segment
  bool hiL = lane >= 16;

  // Padded base: row h <-> kh=0, col w <-> kw=0 (halo absorbs the -1).
  const __bf16* bbase = xc + ((size_t)(b * HP + h) * WP + w) * CIN + (hiL ? 16 : 0);

  v8f acc = {};
  #pragma unroll
  for (int kc = 0; kc < NKC; ++kc) {
    int r   = kc >> 2;
    int kh  = r / 3, kw2 = r - kh * 3;
    int off = (kh * WP + kw2) * CIN + ((kc & 3) << 5);   // compile-time constant
    v16bf bfrag = *(const v16bf*)(bbase + off);
    v16bf afrag = *(const v16bf*)(&Alds[kc * 512 + lane * 16]);
    acc = __builtin_amdgcn_wmma_f32_16x16x32_bf16(
        false, afrag, false, bfrag, (short)0, acc, false, false);
  }

  if (w < W_) {
    int mbase = mt * 16 + (hiL ? 8 : 0);
    float* obase = out + (((size_t)b * COUT + mbase) * H_ + h) * W_ + w;
    #pragma unroll
    for (int r8 = 0; r8 < 8; ++r8)
      obase[(size_t)r8 * H_ * W_] = acc[r8] + bq[mbase + r8];
  }
}

// ---------------- Kernels 6/7: deterministic per-sample stats ----------------
__global__ void k_stat1(const float* __restrict__ z, float* __restrict__ part) {
  int b = blockIdx.x / PB, p = blockIdx.x % PB;
  const float* base = z + (size_t)b * NB + (size_t)p * 8192;
  float s = 0.f, s2 = 0.f;
  for (int i = threadIdx.x; i < 8192; i += 256) { float v = base[i]; s += v; s2 += v * v; }
  __shared__ float sm[256], sq[256];
  sm[threadIdx.x] = s; sq[threadIdx.x] = s2; __syncthreads();
  for (int o = 128; o > 0; o >>= 1) {
    if (threadIdx.x < o) { sm[threadIdx.x] += sm[threadIdx.x + o]; sq[threadIdx.x] += sq[threadIdx.x + o]; }
    __syncthreads();
  }
  if (threadIdx.x == 0) { part[(b * PB + p) * 2] = sm[0]; part[(b * PB + p) * 2 + 1] = sq[0]; }
}

__global__ void k_stat2(const float* __restrict__ part, float* __restrict__ bstats) {
  int b = blockIdx.x, t = threadIdx.x;
  __shared__ float sm[128], sq[128];
  sm[t] = (t < PB) ? part[(b * PB + t) * 2]     : 0.f;
  sq[t] = (t < PB) ? part[(b * PB + t) * 2 + 1] : 0.f;
  __syncthreads();
  for (int o = 64; o > 0; o >>= 1) {
    if (t < o) { sm[t] += sm[t + o]; sq[t] += sq[t + o]; }
    __syncthreads();
  }
  if (t == 0) {
    float nf   = (float)NB;
    float mean = sm[0] / nf;
    float var  = (sq[0] - sm[0] * sm[0] / nf) / (nf - 1.f);
    float inv  = 1.f / (sqrtf(var) + 1e-5f);
    bstats[b * 2] = mean; bstats[b * 2 + 1] = inv;
  }
}

// ---------------- Kernel 8: normalize + ReLU in place ----------------
__global__ void k_norm(float* __restrict__ z, const float* __restrict__ bstats) {
  size_t total = (size_t)B_ * NB;
  for (size_t g = (size_t)blockIdx.x * 256 + threadIdx.x; g < total;
       g += (size_t)gridDim.x * 256) {
    int b = (int)(g / NB);
    float m = bstats[b * 2], inv = bstats[b * 2 + 1];
    float v = (z[g] - m) * inv;
    z[g] = v > 0.f ? v : 0.f;
  }
}

// ---------------- Launcher ----------------
extern "C" void kernel_launch(void* const* d_in, const int* in_sizes, int n_in,
                              void* d_out, int out_size, void* d_ws, size_t ws_size,
                              hipStream_t stream) {
  const float* x    = (const float*)d_in[0];
  const float* w    = (const float*)d_in[1];
  const float* bias = (const float*)d_in[2];
  float* out = (float*)d_out;

  char* ws = (char*)d_ws;
  float* misc   = (float*)ws;          // [0]=meanW [1]=meanB
  float* bq     = misc + 16;           // 256 floats
  float* bstats = bq + 256;            // 64 floats
  float* part   = bstats + 64;         // 32*98*2 floats
  __bf16* Apack = (__bf16*)(ws + (64 << 10));   // 589,824 B
  __bf16* xc    = (__bf16*)(ws + (1 << 20));    // padded NHWC + slack

  k_means<<<1, 1024, 0, stream>>>(w, bias, misc);
  k_packA<<<16 * NKC, 32, 0, stream>>>(w, misc, Apack);
  k_bq   <<<1, 256, 0, stream>>>(bias, misc, bq);
  k_zero <<<2048, 256, 0, stream>>>((uint4*)xc, XC_BYTES / 16);
  k_xc   <<<(B_ * H_ * W_ * CIN) / 256, 256, 0, stream>>>(x, xc);
  k_conv <<<32 * 28 * 16, 256, 0, stream>>>(Apack, xc, bq, out);
  k_stat1<<<32 * PB, 256, 0, stream>>>(out, part);
  k_stat2<<<32, 128, 0, stream>>>(part, bstats);
  k_norm <<<4096, 256, 0, stream>>>(out, bstats);
}
